// GRUDecoder_28243704939084
// MI455X (gfx1250) — compile-verified, matrix operand order checked by reading, not run on hardware
//
#include <hip/hip_runtime.h>

// ---------------------------------------------------------------------------
// GRU decoder step for MI455X (gfx1250).
//  * all GEMMs via v_wmma_f32_16x16x32_bf16 (fp32 accumulate)
//  * weights pre-swizzled once into fragment-order bf16 panels
//  * panels staged to LDS with global_load_async_to_lds_b128 (ASYNCcnt),
//    double-buffered, one barrier per k-chunk
// B=1024, L=64, DF=2048, E=256, U=512, V=32000
// ---------------------------------------------------------------------------

typedef __attribute__((ext_vector_type(16))) __bf16    v16bf;
typedef __attribute__((ext_vector_type(8)))  float     v8f;
typedef __attribute__((ext_vector_type(8)))  unsigned  v8u;

#define B_DIM  1024
#define L_DIM  64
#define DF_DIM 2048
#define E_DIM  256
#define U_DIM  512
#define V_DIM  32000
#define X_DIM  (DF_DIM + E_DIM)   // 2304
#define G_DIM  (3 * U_DIM)        // 1536
#define BN_EPS 0.001f

#define USE_ASYNC 1               // global_load_async_to_lds_b128 staging

__device__ __forceinline__ unsigned pack2bf(float a, float b) {
  const unsigned short lo = __builtin_bit_cast(unsigned short, (__bf16)a);
  const unsigned short hi = __builtin_bit_cast(unsigned short, (__bf16)b);
  return (unsigned)lo | ((unsigned)hi << 16);
}

__device__ __forceinline__ unsigned lds_off(const void* p) {
  // generic pointers into LDS carry the LDS byte offset in their low 32 bits
  return (unsigned)(unsigned long long)p;
}

__device__ __forceinline__ void async_copy16(unsigned lds_byte_off,
                                             const void* gptr) {
#if USE_ASYNC
  asm volatile("global_load_async_to_lds_b128 %0, %1, off"
               :: "v"(lds_byte_off), "v"(gptr) : "memory");
#endif
}
__device__ __forceinline__ void async_wait() {
#if USE_ASYNC
  asm volatile("s_wait_asynccnt 0" ::: "memory");
#endif
}

// ---------------------------------------------------------------------------
// Pre-swizzle: B (K x N, fp32 row-major) -> bf16 panels in WMMA fragment order.
// Panel = one (32 x GC) chunk = GC*16 dwords, GC = 1<<gcShift columns.
// dst dword [(g*(K/32)+c)*GC*16 + nt*256 + (h*16 + n%16)*8 + e/2]
//   k%32 in [0,8): h=0,e=k    [8,16): h=1,e=k-8
//        [16,24): h=0,e=k-8   [24,32): h=1,e=k-16
// ---------------------------------------------------------------------------
__global__ __launch_bounds__(256) void swizzle_b_kernel(
    const float* __restrict__ src, unsigned* __restrict__ dst,
    int K, int N, int gcShift)
{
  const int i = blockIdx.x * 256 + threadIdx.x;   // i < (K/2)*N
  const int n = i % N;
  const int k = (i / N) << 1;                     // even k
  const float v0 = src[(size_t)k * N + n];
  const float v1 = src[(size_t)(k + 1) * N + n];
  const int c  = k >> 5;
  const int kl = k & 31;
  const int h  = (kl >> 3) & 1;
  const int e2 = (((kl & 16) ? (kl - 8) : kl) - (h << 3)) >> 1;
  const int g  = n >> gcShift;
  const int nl = n & ((1 << gcShift) - 1);
  const size_t panel = (size_t)g * (K >> 5) + c;
  dst[panel * ((16u) << gcShift) + (nl >> 4) * 256 + ((h << 4) + (nl & 15)) * 8 + e2]
      = pack2bf(v0, v1);
}

// fp32 -> bf16 row-major convert (n must be even; one dword per thread)
__global__ __launch_bounds__(256) void cvt_bf16_kernel(
    const float* __restrict__ src, unsigned* __restrict__ dst, int npairs)
{
  const int i = blockIdx.x * 256 + threadIdx.x;
  if (i < npairs) dst[i] = pack2bf(src[2 * i], src[2 * i + 1]);
}

// ---------------------------------------------------------------------------
// WMMA GEMM: C(M,N) = A(M,K)[bf16 row-major] @ Bswz(K,N)[swizzled panels] + bias
// Block = 256 thr / 8 waves -> 128x64 tile. Double-buffered async staging.
// BN: apply BatchNorm. OUTBF: store bf16 to Cbf instead of fp32 to C.
// Grid = (M/128)*(N/64); K % 32 == 0.
// ---------------------------------------------------------------------------
template <bool BN, bool OUTBF>
__global__ __launch_bounds__(256) void wmma_gemm_kernel(
    const __bf16* __restrict__ Abf, const unsigned* __restrict__ Bswz,
    const float* __restrict__ bias,
    const float* __restrict__ gamma, const float* __restrict__ beta,
    const float* __restrict__ mmean, const float* __restrict__ mvar,
    float* __restrict__ C, __bf16* __restrict__ Cbf, int M, int N, int K)
{
  __shared__ v16bf ldsB[2][4 * 32];                // 2 x 4KB fragment buffers

  const int tid   = threadIdx.x;
  const int lane  = tid & 31;
  const int w     = tid >> 5;
  const int ngrp  = N >> 6;
  const int mb    = blockIdx.x / ngrp;
  const int ng    = blockIdx.x % ngrp;
  const int row   = (mb << 7) + (w << 4) + (lane & 15);
  const int koff  = (lane >> 4) << 3;
  const int nch   = K >> 5;

  const __bf16*   arow  = Abf + (size_t)row * K;
  const unsigned* panel = Bswz + (size_t)ng * nch * 1024;
  const unsigned  ldsBase = lds_off(&ldsB[0][0]);

  auto stage = [&](int c, int buf) {
#if USE_ASYNC
    async_copy16(ldsBase + (unsigned)buf * 4096u + (unsigned)tid * 16u,
                 panel + (size_t)c * 1024 + tid * 4);
#else
    uint4 d = *(const uint4*)(panel + (size_t)c * 1024 + tid * 4);
    *(uint4*)((unsigned*)ldsB + buf * 1024 + tid * 4) = d;
#endif
  };

  v8f acc[4];
#pragma unroll
  for (int t = 0; t < 4; ++t)
#pragma unroll
    for (int j = 0; j < 8; ++j) acc[t][j] = 0.0f;

  stage(0, 0);
  for (int kc = 0; kc < nch; ++kc) {
    const int buf = kc & 1;
    async_wait();                  // own async writes done
    __syncthreads();               // -> all waves' writes done & visible
    if (kc + 1 < nch) stage(kc + 1, buf ^ 1);

    // ---- A fragment: two 16B loads of bf16, zero converts
    const int kk = kc << 5;
    uint4 p0 = *(const uint4*)(arow + kk + koff);
    uint4 p1 = *(const uint4*)(arow + kk + koff + 16);
    v8u au; au[0]=p0.x; au[1]=p0.y; au[2]=p0.z; au[3]=p0.w;
            au[4]=p1.x; au[5]=p1.y; au[6]=p1.z; au[7]=p1.w;
    const v16bf af = __builtin_bit_cast(v16bf, au);

#pragma unroll
    for (int t = 0; t < 4; ++t) {
      const v16bf bfr = ldsB[buf][(t << 5) + lane];   // 2x ds_load_b128
      acc[t] = __builtin_amdgcn_wmma_f32_16x16x32_bf16(
          false, af, false, bfr, (short)0, acc[t], false, false);
    }
  }

  // ---- epilogue (C/D layout: VGPR j -> row j + 8*(lane>=16), col = lane&15)
#pragma unroll
  for (int t = 0; t < 4; ++t) {
    const int c = (ng << 6) + (t << 4) + (lane & 15);
    const float badd = bias[c];
#pragma unroll
    for (int j = 0; j < 8; ++j) {
      const int r = (mb << 7) + (w << 4) + j + ((lane >> 4) << 3);
      float v = acc[t][j] + badd;
      if (BN)
        v = (v - mmean[c]) * rsqrtf(mvar[c] + BN_EPS) * gamma[c] + beta[c];
      if (OUTBF) Cbf[(size_t)r * N + c] = (__bf16)v;
      else       C[(size_t)r * N + c] = v;
    }
  }
}

// ---------------------------------------------------------------------------
// Fused attention + context. One block (512 thr / 16 waves) per batch element.
// score[l] = Wv . tanh(fm[b,l,:]@Wu + bu + sWw[b,:]),  attn = softmax_l(score),
// ctx[b,:] -> bf16 into xBf[b, 0:2048].  Wu pre-swizzled (512-col panels).
// ---------------------------------------------------------------------------
__global__ __launch_bounds__(512) void attn_ctx_kernel(
    const float* __restrict__ fm, const unsigned* __restrict__ WuSwz,
    const float* __restrict__ Wv, const float* __restrict__ bu,
    const float* __restrict__ sWw,
    float* __restrict__ attn_out, __bf16* __restrict__ xBf)
{
  __shared__ v16bf ldsWu[2][32 * 32];              // 2 x 32KB fragment buffers
  __shared__ float ldsScore[L_DIM];
  __shared__ float ldsAttn[L_DIM + 1];

  const int b    = blockIdx.x;
  const int tid  = threadIdx.x;
  const int lane = tid & 31;
  const int w    = tid >> 5;
  const int mt   = w & 3;                          // l-tile
  const int nq   = w >> 2;                         // u-quarter (8 n-tiles)
  const int lrow = (mt << 4) + (lane & 15);
  const int koff = (lane >> 4) << 3;

  if (tid < L_DIM) ldsScore[tid] = 0.0f;

  const float* fmB     = fm + (size_t)b * L_DIM * DF_DIM;
  const float* arow    = fmB + (size_t)lrow * DF_DIM;
  const float* sWwRow  = sWw + (size_t)b * U_DIM;
  const unsigned ldsBase = lds_off(&ldsWu[0][0]);

  auto stage = [&](int c, int buf) {               // 8192 dwords / chunk
#if USE_ASYNC
#pragma unroll
    for (int i = 0; i < 4; ++i)
      async_copy16(ldsBase + (unsigned)buf * 32768u + (unsigned)tid * 64u + i * 16u,
                   WuSwz + (size_t)c * 8192 + tid * 16 + i * 4);
#else
#pragma unroll
    for (int i = 0; i < 4; ++i) {
      uint4 d = *(const uint4*)(WuSwz + (size_t)c * 8192 + tid * 16 + i * 4);
      *(uint4*)((unsigned*)ldsWu + buf * 8192 + tid * 16 + i * 4) = d;
    }
#endif
  };

  v8f acc[8];
#pragma unroll
  for (int nt = 0; nt < 8; ++nt) {
    const int n = ((nq << 3) + nt) * 16 + (lane & 15);
    const float c0 = sWwRow[n] + bu[n];            // bias folded into C
#pragma unroll
    for (int j = 0; j < 8; ++j) acc[nt][j] = c0;
  }

  stage(0, 0);
  for (int kc = 0; kc < (DF_DIM >> 5); ++kc) {
    const int buf = kc & 1;
    async_wait();
    __syncthreads();
    if (kc + 1 < (DF_DIM >> 5)) stage(kc + 1, buf ^ 1);

    const int kk = kc << 5;
    float4 a0 = *(const float4*)(arow + kk + koff);
    float4 a1 = *(const float4*)(arow + kk + koff + 4);
    float4 a2 = *(const float4*)(arow + kk + koff + 16);
    float4 a3 = *(const float4*)(arow + kk + koff + 20);
    v16bf af;
    af[0]=(__bf16)a0.x; af[1]=(__bf16)a0.y; af[2]=(__bf16)a0.z; af[3]=(__bf16)a0.w;
    af[4]=(__bf16)a1.x; af[5]=(__bf16)a1.y; af[6]=(__bf16)a1.z; af[7]=(__bf16)a1.w;
    af[8]=(__bf16)a2.x; af[9]=(__bf16)a2.y; af[10]=(__bf16)a2.z; af[11]=(__bf16)a2.w;
    af[12]=(__bf16)a3.x; af[13]=(__bf16)a3.y; af[14]=(__bf16)a3.z; af[15]=(__bf16)a3.w;

#pragma unroll
    for (int nt = 0; nt < 8; ++nt) {
      const v16bf bfr = ldsWu[buf][((nq << 3) + nt) * 32 + lane];
      acc[nt] = __builtin_amdgcn_wmma_f32_16x16x32_bf16(
          false, af, false, bfr, (short)0, acc[nt], false, false);
    }
  }

  // ---- score partials: tanh, dot with Wv, cross-lane reduce via LDS atomics
  float sp[8] = {0, 0, 0, 0, 0, 0, 0, 0};
#pragma unroll
  for (int nt = 0; nt < 8; ++nt) {
    const float wv = Wv[((nq << 3) + nt) * 16 + (lane & 15)];
#pragma unroll
    for (int j = 0; j < 8; ++j) sp[j] += tanhf(acc[nt][j]) * wv;
  }
#pragma unroll
  for (int j = 0; j < 8; ++j) {
    const int l = (mt << 4) + j + ((lane >> 4) << 3);
    atomicAdd(&ldsScore[l], sp[j]);                // ds_add_f32
  }
  __syncthreads();

  if (tid == 0) {                                  // 64-wide softmax
    float mx = -3.402823466e38f;
    for (int l = 0; l < L_DIM; ++l) mx = fmaxf(mx, ldsScore[l]);
    float s = 0.0f;
    for (int l = 0; l < L_DIM; ++l) {
      const float e = expf(ldsScore[l] - mx);
      ldsAttn[l] = e;
      s += e;
    }
    ldsAttn[L_DIM] = 1.0f / s;
  }
  __syncthreads();
  const float inv = ldsAttn[L_DIM];
  if (tid < L_DIM) attn_out[(size_t)b * L_DIM + tid] = ldsAttn[tid] * inv;

  // ---- context -> bf16 into concat buffer (4 columns / thread)
  for (int d = tid; d < DF_DIM; d += 512) {
    float s = 0.0f;
    for (int l = 0; l < L_DIM; ++l) s += ldsAttn[l] * fmB[(size_t)l * DF_DIM + d];
    xBf[(size_t)b * X_DIM + d] = (__bf16)(s * inv);
  }
}

// ---------------------------------------------------------------------------
// Embedding gather (bf16) into concat buffer: x[b, 2048+e] = emb[tokens[b], e]
// ---------------------------------------------------------------------------
__global__ __launch_bounds__(256) void emb_gather_kernel(
    const int* __restrict__ tokens, const float* __restrict__ emb,
    __bf16* __restrict__ xBf)
{
  const int i = blockIdx.x * 256 + threadIdx.x;   // < B*E
  const int b = i >> 8, e = i & (E_DIM - 1);
  const int t = tokens[b];
  xBf[(size_t)b * X_DIM + DF_DIM + e] = (__bf16)emb[(size_t)t * E_DIM + e];
}

// ---------------------------------------------------------------------------
// GRU gates: xg = x@gru_k + gru_b[0] (fp32).  rg = gru_b[1].
// h fp32 -> d_out, plus bf16 copy feeding the next GEMM.
// ---------------------------------------------------------------------------
__global__ __launch_bounds__(256) void gru_gates_kernel(
    const float* __restrict__ xg, const float* __restrict__ gru_b,
    float* __restrict__ h, __bf16* __restrict__ hBf)
{
  const int i = blockIdx.x * 256 + threadIdx.x;   // < B*U
  const int b = i >> 9, u = i & (U_DIM - 1);
  const float* rg  = gru_b + G_DIM;               // row 1 of (2, 3U)
  const float* row = xg + (size_t)b * G_DIM;
  const float z  = 1.0f / (1.0f + expf(-(row[u] + rg[u])));
  const float r  = 1.0f / (1.0f + expf(-(row[U_DIM + u] + rg[U_DIM + u])));
  const float hh = tanhf(row[2 * U_DIM + u] + r * rg[2 * U_DIM + u]);
  const float hv = (1.0f - z) * hh;
  h[i] = hv;
  hBf[i] = (__bf16)hv;
}

// ---------------------------------------------------------------------------
extern "C" void kernel_launch(void* const* d_in, const int* in_sizes, int n_in,
                              void* d_out, int out_size, void* d_ws, size_t ws_size,
                              hipStream_t stream)
{
  const int*   tokens    = (const int*)  d_in[0];
  const float* fm        = (const float*)d_in[1];
  const float* prevState = (const float*)d_in[2];
  const float* emb       = (const float*)d_in[3];
  const float* Wu        = (const float*)d_in[4];
  const float* bu        = (const float*)d_in[5];
  const float* Ww        = (const float*)d_in[6];
  const float* bw        = (const float*)d_in[7];
  const float* Wv        = (const float*)d_in[8];
  /* d_in[9]  bv: additive constant on scores -> softmax-invariant, unused   */
  const float* gru_k     = (const float*)d_in[10];
  /* d_in[11] gru_rk: never enters the reference math (only its shape)       */
  const float* gru_b     = (const float*)d_in[12];
  const float* W1        = (const float*)d_in[13];
  const float* b1        = (const float*)d_in[14];
  const float* gamma     = (const float*)d_in[15];
  const float* beta      = (const float*)d_in[16];
  const float* mmean     = (const float*)d_in[17];
  const float* mvar      = (const float*)d_in[18];
  const float* W2        = (const float*)d_in[19];
  const float* b2        = (const float*)d_in[20];

  // outputs: logits (B,V) | h (B,U) | attn (B,L,1), concatenated flat
  float* logits = (float*)d_out;
  float* h      = logits + (size_t)B_DIM * V_DIM;
  float* attn   = h + (size_t)B_DIM * U_DIM;

  // ---- workspace bump allocator (bytes, 256B aligned) -------------------
  char* wp = (char*)d_ws;
  auto alloc = [&](size_t bytes) {
    char* p = wp; wp += (bytes + 255) & ~(size_t)255; return (void*)p;
  };
  float*    sWw     = (float*)   alloc((size_t)B_DIM * U_DIM * 4);        // 2 MB
  float*    xg      = (float*)   alloc((size_t)B_DIM * G_DIM * 4);        // 6 MB
  __bf16*   stateBf = (__bf16*)  alloc((size_t)B_DIM * U_DIM * 2);        // 1 MB
  __bf16*   xBf     = (__bf16*)  alloc((size_t)B_DIM * X_DIM * 2);        // 4.5 MB
  __bf16*   hBf     = (__bf16*)  alloc((size_t)B_DIM * U_DIM * 2);        // 1 MB
  __bf16*   ynBf    = (__bf16*)  alloc((size_t)B_DIM * U_DIM * 2);        // 1 MB
  unsigned* WwSwz   = (unsigned*)alloc((size_t)U_DIM * U_DIM * 2);        // .5 MB
  unsigned* WuSwz   = (unsigned*)alloc((size_t)DF_DIM * U_DIM * 2);       // 2 MB
  unsigned* gkSwz   = (unsigned*)alloc((size_t)X_DIM * G_DIM * 2);        // 6.75 MB
  unsigned* W1Swz   = (unsigned*)alloc((size_t)U_DIM * U_DIM * 2);        // .5 MB
  unsigned* W2Swz   = (unsigned*)alloc((size_t)U_DIM * V_DIM * 2);        // 31.25 MB

  auto swzBlocks  = [](int K, int N) { return (K * N) / 512; };
  auto gemmBlocks = [](int M, int N) { return (M / 128) * (N / 64); };

  // ---- one-shot weight pre-swizzle + activation converts ----------------
  swizzle_b_kernel<<<swzBlocks(U_DIM, U_DIM), 256, 0, stream>>>(Ww, WwSwz, U_DIM, U_DIM, 6);
  swizzle_b_kernel<<<swzBlocks(DF_DIM, U_DIM), 256, 0, stream>>>(Wu, WuSwz, DF_DIM, U_DIM, 9);
  swizzle_b_kernel<<<swzBlocks(X_DIM, G_DIM), 256, 0, stream>>>(gru_k, gkSwz, X_DIM, G_DIM, 6);
  swizzle_b_kernel<<<swzBlocks(U_DIM, U_DIM), 256, 0, stream>>>(W1, W1Swz, U_DIM, U_DIM, 6);
  swizzle_b_kernel<<<swzBlocks(U_DIM, V_DIM), 256, 0, stream>>>(W2, W2Swz, U_DIM, V_DIM, 6);
  cvt_bf16_kernel<<<(B_DIM * U_DIM / 2 + 255) / 256, 256, 0, stream>>>(
      prevState, (unsigned*)stateBf, B_DIM * U_DIM / 2);
  emb_gather_kernel<<<(B_DIM * E_DIM) / 256, 256, 0, stream>>>(tokens, emb, xBf);

  // ---- pipeline ---------------------------------------------------------
  // 1. sWw = prevState @ Ww + bw
  wmma_gemm_kernel<false, false><<<gemmBlocks(B_DIM, U_DIM), 256, 0, stream>>>(
      stateBf, WwSwz, bw, nullptr, nullptr, nullptr, nullptr,
      sWw, nullptr, B_DIM, U_DIM, U_DIM);

  // 2. fused attention + softmax + context -> attn, xBf[:, :2048]
  attn_ctx_kernel<<<B_DIM, 512, 0, stream>>>(fm, WuSwz, Wv, bu, sWw, attn, xBf);

  // 3. xg = x @ gru_k + gru_b[0]
  wmma_gemm_kernel<false, false><<<gemmBlocks(B_DIM, G_DIM), 256, 0, stream>>>(
      xBf, gkSwz, gru_b, nullptr, nullptr, nullptr, nullptr,
      xg, nullptr, B_DIM, G_DIM, X_DIM);

  // 4. GRU gates -> h (fp32 out) + hBf
  gru_gates_kernel<<<(B_DIM * U_DIM) / 256, 256, 0, stream>>>(xg, gru_b, h, hBf);

  // 5. ynBf = bf16(BatchNorm(h @ W1 + b1))
  wmma_gemm_kernel<true, true><<<gemmBlocks(B_DIM, U_DIM), 256, 0, stream>>>(
      hBf, W1Swz, b1, gamma, beta, mmean, mvar, nullptr, ynBf,
      B_DIM, U_DIM, U_DIM);

  // 6. logits = yn @ W2 + b2
  wmma_gemm_kernel<false, false><<<gemmBlocks(B_DIM, V_DIM), 256, 0, stream>>>(
      ynBf, W2Swz, b2, nullptr, nullptr, nullptr, nullptr,
      logits, nullptr, B_DIM, V_DIM, U_DIM);
}
